// LoRAAttnProcessor_45896020525754
// MI455X (gfx1250) — matrix operand order repaired and checked by reading, not compiled, not run on hardware
//
#include <hip/hip_runtime.h>
#include <hip/hip_bf16.h>

typedef __attribute__((ext_vector_type(16))) _Float16 v16h;
typedef __attribute__((ext_vector_type(8)))  float    v8f;
typedef __attribute__((ext_vector_type(4)))  int      i4;

#define C_DIM 1280
#define S_DIM 2048
#define B_DIM 2
#define H_DIM 20
#define D_DIM 64
#define R_DIM 4
#define M_TOT (B_DIM * S_DIM)   // 4096

union HF { v16h v; _Float16 h[16]; };

using pkh2 = decltype(__builtin_amdgcn_cvt_pkrtz(0.0f, 0.0f));
union H4 { pkh2 p[2]; unsigned long long u; };

// ---- async global->LDS copy (128-bit), with safe fallback --------------------
#if __has_builtin(__builtin_amdgcn_global_load_async_to_lds_b128) && \
    __has_builtin(__builtin_amdgcn_s_wait_asynccnt)
#define AS1 __attribute__((address_space(1)))
#define AS3 __attribute__((address_space(3)))
__device__ inline void cp_async16(void* lds, const void* g) {
  __builtin_amdgcn_global_load_async_to_lds_b128(
      (AS1 i4*)(void*)g, (AS3 i4*)lds, 0, 0);
}
__device__ inline void cp_wait() { __builtin_amdgcn_s_wait_asynccnt(0); }
#else
__device__ inline void cp_async16(void* lds, const void* g) {
  *(uint4*)lds = *(const uint4*)g;
}
__device__ inline void cp_wait() {}
#endif

__device__ inline v8f wmma_f16(v16h a, v16h b, v8f c) {
  // D = A(16x32 f16) * B(32x16 f16) + C(16x16 f32)
  return __builtin_amdgcn_wmma_f32_16x16x32_f16(
      false, a, false, b, (short)0, c, false, false);
}

// A-matrix fragment (16x32 f16) from LDS tile, leading dim LD (halves).
// lane l holds row M = l%16; element e -> K = 16*(l/16) + e
__device__ inline v16h ldsA(const _Float16* t, int LD, int lane) {
  HF f;
  int g = lane >> 4, r = lane & 15;
#pragma unroll
  for (int e = 0; e < 16; ++e) {
    int kk = g * 16 + e;
    f.h[e] = t[r * LD + kk];
  }
  return f.v;
}

// B fragment (32x16), tile stored [n][k] row-major: value B[k][n] = tile[n][k]
__device__ inline v16h ldsB_nk(const _Float16* t, int LD, int lane) {
  HF f;
  int g = lane >> 4, n = lane & 15;
#pragma unroll
  for (int e = 0; e < 16; ++e) f.h[e] = t[n * LD + g * 16 + e];
  return f.v;
}

// B fragment (32x16), tile stored [k][n] row-major (V-tile)
__device__ inline v16h ldsB_kn(const _Float16* t, int LD, int lane) {
  HF f;
  int g = lane >> 4, n = lane & 15;
#pragma unroll
  for (int e = 0; e < 16; ++e) f.h[e] = t[(g * 16 + e) * LD + n];
  return f.v;
}

// ---------------------------------------------------------------------------
// Kernel 0: W_eff[n][c] = W[n][c] + sum_r Bm[n][r]*A[r][c]   (rank-4 LoRA fold)
// one thread = 4 consecutive c  (float4 loads, packed f16 store)
// ---------------------------------------------------------------------------
__global__ void fold_lora(const float* __restrict__ W,
                          const float* __restrict__ A,   // [R, C]
                          const float* __restrict__ Bm,  // [C, R]
                          _Float16* __restrict__ We) {
  int idx = blockIdx.x * blockDim.x + threadIdx.x;   // over C*C/4
  if (idx >= C_DIM * C_DIM / 4) return;
  int n = idx / (C_DIM / 4), c4 = idx % (C_DIM / 4);
  float4 w = *(const float4*)&W[n * C_DIM + c4 * 4];
  float b0 = Bm[n * R_DIM + 0], b1 = Bm[n * R_DIM + 1];
  float b2 = Bm[n * R_DIM + 2], b3 = Bm[n * R_DIM + 3];
  float4 a0 = *(const float4*)&A[0 * C_DIM + c4 * 4];
  float4 a1 = *(const float4*)&A[1 * C_DIM + c4 * 4];
  float4 a2 = *(const float4*)&A[2 * C_DIM + c4 * 4];
  float4 a3 = *(const float4*)&A[3 * C_DIM + c4 * 4];
  w.x += b0 * a0.x + b1 * a1.x + b2 * a2.x + b3 * a3.x;
  w.y += b0 * a0.y + b1 * a1.y + b2 * a2.y + b3 * a3.y;
  w.z += b0 * a0.z + b1 * a1.z + b2 * a2.z + b3 * a3.z;
  w.w += b0 * a0.w + b1 * a1.w + b2 * a2.w + b3 * a3.w;
  H4 p;
  p.p[0] = __builtin_amdgcn_cvt_pkrtz(w.x, w.y);
  p.p[1] = __builtin_amdgcn_cvt_pkrtz(w.z, w.w);
  *(unsigned long long*)&We[n * C_DIM + c4 * 4] = p.u;
}

// ---------------------------------------------------------------------------
// Kernel 1: fused q/k/v projection.  y = x @ W_eff^T, stored f16 in [B,H,S,D].
// grid = (M/64, H, 3); block = 128 (4 waves).  q pre-scaled by 1/sqrt(D).
// ---------------------------------------------------------------------------
__global__ void __launch_bounds__(128)
qkv_proj(const float* __restrict__ x,
         const _Float16* __restrict__ Wq, const _Float16* __restrict__ Wk,
         const _Float16* __restrict__ Wv,
         _Float16* __restrict__ qo, _Float16* __restrict__ ko,
         _Float16* __restrict__ vo) {
  __shared__ _Float16 Xs[64 * 32];
  __shared__ _Float16 Ws[64 * 32];

  int z = blockIdx.z;
  const _Float16* W = (z == 0) ? Wq : (z == 1) ? Wk : Wv;
  _Float16* out     = (z == 0) ? qo : (z == 1) ? ko : vo;
  float scale       = (z == 0) ? 0.125f : 1.0f;

  int m0 = blockIdx.x * 64;
  int h  = blockIdx.y;           // head == 64-wide column block
  int n0 = h * 64;
  int tid = threadIdx.x, lane = tid & 31, wave = tid >> 5;

  v8f acc[4] = {};

  for (int k0 = 0; k0 < C_DIM; k0 += 32) {
    // X tile 64x32: f32 -> f16 via packed cvt; 512 float4, 4 per thread
#pragma unroll
    for (int t = 0; t < 4; ++t) {
      int j = tid + t * 128;            // 0..511
      int r = j >> 3, c4 = j & 7;
      float4 f = *(const float4*)&x[(m0 + r) * C_DIM + k0 + c4 * 4];
      H4 p;
      p.p[0] = __builtin_amdgcn_cvt_pkrtz(f.x, f.y);
      p.p[1] = __builtin_amdgcn_cvt_pkrtz(f.z, f.w);
      *(unsigned long long*)&Xs[r * 32 + c4 * 4] = p.u;
    }
    // W tile 64x32 f16: 128 x 128-bit async copies, 1 per thread
    {
      int r = tid >> 2, c8 = tid & 3;
      cp_async16(&Ws[r * 32 + c8 * 8], &W[(n0 + r) * C_DIM + k0 + c8 * 8]);
    }
    cp_wait();
    __syncthreads();

    v16h a = ldsA(Xs + wave * 16 * 32, 32, lane);
#pragma unroll
    for (int n = 0; n < 4; ++n) {
      v16h b = ldsB_nk(Ws + n * 16 * 32, 32, lane);
      acc[n] = wmma_f16(a, b, acc[n]);
    }
    __syncthreads();
  }

  // scatter to [B,H,S,D] f16
  int g = lane >> 4, cl = lane & 15;
#pragma unroll
  for (int n = 0; n < 4; ++n)
#pragma unroll
    for (int e = 0; e < 8; ++e) {
      int m = m0 + wave * 16 + e + 8 * g;      // token index in [0, B*S)
      int b = m >> 11, s = m & (S_DIM - 1);
      int d = n * 16 + cl;
      out[(((b * H_DIM + h) * S_DIM + s) * D_DIM) + d] =
          (_Float16)(acc[n][e] * scale);
    }
}

// ---------------------------------------------------------------------------
// Kernel 2: flash attention.  grid = (S/64, B*H); block = 128 (4 waves).
// q already scaled by 1/sqrt(D).  Writes ctx f16 to [B,S,C].
// ---------------------------------------------------------------------------
__global__ void __launch_bounds__(128)
attn(const _Float16* __restrict__ q, const _Float16* __restrict__ k,
     const _Float16* __restrict__ v, _Float16* __restrict__ ctx) {
  __shared__ _Float16 Qs[64 * 64];
  __shared__ _Float16 Kt[64 * 64];
  __shared__ _Float16 Vt[64 * 64];
  __shared__ _Float16 Ps[4][16 * 64];

  int qb = blockIdx.x;           // query block (64 rows)
  int bh = blockIdx.y;           // b*H + h
  const _Float16* qbh = q + bh * S_DIM * D_DIM;
  const _Float16* kbh = k + bh * S_DIM * D_DIM;
  const _Float16* vbh = v + bh * S_DIM * D_DIM;

  int tid = threadIdx.x, lane = tid & 31, wave = tid >> 5;
  int g = lane >> 4, cl = lane & 15;

  // Q tile 64x64 f16: 512 x 128-bit copies, 4 per thread
#pragma unroll
  for (int t = 0; t < 4; ++t) {
    int j = tid + t * 128;
    cp_async16(&Qs[j * 8], &qbh[qb * 64 * D_DIM + j * 8]);
  }
  cp_wait();
  __syncthreads();

  v16h qa0 = ldsA(Qs + wave * 16 * 64, 64, lane);       // d 0..31
  v16h qa1 = ldsA(Qs + wave * 16 * 64 + 32, 64, lane);  // d 32..63

  v8f acc[4] = {};
  float mrow[8], lrow[8];
#pragma unroll
  for (int e = 0; e < 8; ++e) { mrow[e] = -1e30f; lrow[e] = 0.0f; }

  for (int kt = 0; kt < S_DIM; kt += 64) {
#pragma unroll
    for (int t = 0; t < 4; ++t) {
      int j = tid + t * 128;
      cp_async16(&Kt[j * 8], &kbh[kt * D_DIM + j * 8]);
      cp_async16(&Vt[j * 8], &vbh[kt * D_DIM + j * 8]);
    }
    cp_wait();
    __syncthreads();

    // scores = Q x K^T  (16 rows per wave, 64 keys)
    v8f sc[4] = {};
#pragma unroll
    for (int n = 0; n < 4; ++n) {
      sc[n] = wmma_f16(qa0, ldsB_nk(Kt + n * 16 * 64, 64, lane), sc[n]);
      sc[n] = wmma_f16(qa1, ldsB_nk(Kt + n * 16 * 64 + 32, 64, lane), sc[n]);
    }

    // online softmax: row (e + 8*g) spans the lane's 16-lane group
#pragma unroll
    for (int e = 0; e < 8; ++e) {
      float mx = fmaxf(fmaxf(sc[0][e], sc[1][e]), fmaxf(sc[2][e], sc[3][e]));
#pragma unroll
      for (int off = 1; off < 16; off <<= 1)
        mx = fmaxf(mx, __shfl_xor(mx, off, 32));
      float mnew  = fmaxf(mrow[e], mx);
      float alpha = __expf(mrow[e] - mnew);
      mrow[e] = mnew;
      float rs = 0.0f;
#pragma unroll
      for (int n = 0; n < 4; ++n) {
        float p = __expf(sc[n][e] - mnew);
        sc[n][e] = p;
        rs += p;
      }
#pragma unroll
      for (int off = 1; off < 16; off <<= 1)
        rs += __shfl_xor(rs, off, 32);
      lrow[e] = lrow[e] * alpha + rs;
#pragma unroll
      for (int n = 0; n < 4; ++n) acc[n][e] *= alpha;
    }

    // C-layout -> A-layout reshape of P through LDS
#pragma unroll
    for (int n = 0; n < 4; ++n)
#pragma unroll
      for (int e = 0; e < 8; ++e)
        Ps[wave][(e + 8 * g) * 64 + n * 16 + cl] = (_Float16)sc[n][e];
    __syncthreads();

    // O += P x V
#pragma unroll
    for (int c = 0; c < 2; ++c) {
      v16h pa = ldsA(Ps[wave] + 32 * c, 64, lane);
#pragma unroll
      for (int n = 0; n < 4; ++n) {
        v16h vb = ldsB_kn(Vt + (32 * c) * 64 + n * 16, 64, lane);
        acc[n] = wmma_f16(pa, vb, acc[n]);
      }
    }
    __syncthreads();
  }

  int h = bh % H_DIM, b = bh / H_DIM;
#pragma unroll
  for (int e = 0; e < 8; ++e) {
    float invl = 1.0f / lrow[e];
    int s = qb * 64 + wave * 16 + e + 8 * g;
#pragma unroll
    for (int n = 0; n < 4; ++n) {
      int d = n * 16 + cl;
      ctx[(b * S_DIM + s) * C_DIM + h * D_DIM + d] = (_Float16)(acc[n][e] * invl);
    }
  }
}

// ---------------------------------------------------------------------------
// Kernel 3: out = ctx @ Wo_eff^T + bo, fp32 output.  grid = (M/64, C/64).
// ---------------------------------------------------------------------------
__global__ void __launch_bounds__(128)
out_proj(const _Float16* __restrict__ ctx, const _Float16* __restrict__ Wo,
         const float* __restrict__ bias, float* __restrict__ out) {
  __shared__ _Float16 Xs[64 * 32];
  __shared__ _Float16 Ws[64 * 32];

  int m0 = blockIdx.x * 64;
  int n0 = blockIdx.y * 64;
  int tid = threadIdx.x, lane = tid & 31, wave = tid >> 5;

  v8f acc[4] = {};

  for (int k0 = 0; k0 < C_DIM; k0 += 32) {
    // ctx tile 64x32 f16: 256 x 128-bit copies, 2 per thread
#pragma unroll
    for (int t = 0; t < 2; ++t) {
      int j = tid + t * 128;
      int r = j >> 2, c8 = j & 3;
      cp_async16(&Xs[r * 32 + c8 * 8], &ctx[(m0 + r) * C_DIM + k0 + c8 * 8]);
    }
    {
      int r = tid >> 2, c8 = tid & 3;
      cp_async16(&Ws[r * 32 + c8 * 8], &Wo[(n0 + r) * C_DIM + k0 + c8 * 8]);
    }
    cp_wait();
    __syncthreads();

    v16h a = ldsA(Xs + wave * 16 * 32, 32, lane);
#pragma unroll
    for (int n = 0; n < 4; ++n) {
      v16h b = ldsB_nk(Ws + n * 16 * 32, 32, lane);
      acc[n] = wmma_f16(a, b, acc[n]);
    }
    __syncthreads();
  }

  int g = lane >> 4, cl = lane & 15;
#pragma unroll
  for (int n = 0; n < 4; ++n)
#pragma unroll
    for (int e = 0; e < 8; ++e) {
      int m = m0 + wave * 16 + e + 8 * g;
      int nc = n0 + n * 16 + cl;
      out[m * C_DIM + nc] = acc[n][e] + bias[nc];
    }
}

// ---------------------------------------------------------------------------
extern "C" void kernel_launch(void* const* d_in, const int* in_sizes, int n_in,
                              void* d_out, int out_size, void* d_ws, size_t ws_size,
                              hipStream_t stream) {
  const float* x  = (const float*)d_in[0];
  const float* Wq = (const float*)d_in[1];
  const float* Wk = (const float*)d_in[2];
  const float* Wv = (const float*)d_in[3];
  const float* Wo = (const float*)d_in[4];
  const float* bo = (const float*)d_in[5];
  const float* Aq = (const float*)d_in[6];
  const float* Bq = (const float*)d_in[7];
  const float* Ak = (const float*)d_in[8];
  const float* Bk = (const float*)d_in[9];
  const float* Av = (const float*)d_in[10];
  const float* Bv = (const float*)d_in[11];
  const float* Ao = (const float*)d_in[12];
  const float* Bo = (const float*)d_in[13];
  float* out = (float*)d_out;

  // workspace carve-up (f16 buffers)
  const size_t WSZ = (size_t)C_DIM * C_DIM;      // elems per weight
  const size_t TSZ = (size_t)M_TOT * C_DIM;      // elems per activation tensor
  char* ws = (char*)d_ws;
  _Float16* WqE = (_Float16*)ws;                 ws += WSZ * 2;
  _Float16* WkE = (_Float16*)ws;                 ws += WSZ * 2;
  _Float16* WvE = (_Float16*)ws;                 ws += WSZ * 2;
  _Float16* WoE = (_Float16*)ws;                 ws += WSZ * 2;
  _Float16* qb  = (_Float16*)ws;                 ws += TSZ * 2;
  _Float16* kb  = (_Float16*)ws;                 ws += TSZ * 2;
  _Float16* vb  = (_Float16*)ws;                 ws += TSZ * 2;
  _Float16* ctx = (_Float16*)ws;                 ws += TSZ * 2;

  // 0) fold LoRA into weights (f32 -> f16 effective weights)
  dim3 fg((C_DIM * C_DIM / 4 + 255) / 256);
  fold_lora<<<fg, 256, 0, stream>>>(Wq, Aq, Bq, WqE);
  fold_lora<<<fg, 256, 0, stream>>>(Wk, Ak, Bk, WkE);
  fold_lora<<<fg, 256, 0, stream>>>(Wv, Av, Bv, WvE);
  fold_lora<<<fg, 256, 0, stream>>>(Wo, Ao, Bo, WoE);

  // 1) q/k/v projections (q pre-scaled by 1/sqrt(D))
  qkv_proj<<<dim3(M_TOT / 64, H_DIM, 3), 128, 0, stream>>>(
      x, WqE, WkE, WvE, qb, kb, vb);

  // 2) flash attention -> ctx [B,S,C] f16
  attn<<<dim3(S_DIM / 64, B_DIM * H_DIM), 128, 0, stream>>>(qb, kb, vb, ctx);

  // 3) output projection + bias -> fp32 out
  out_proj<<<dim3(M_TOT / 64, C_DIM / 64), 128, 0, stream>>>(ctx, WoE, bo, out);
}
